// CTSelectiveSSMBlock_71219147702390
// MI455X (gfx1250) — matrix operand order repaired and checked by reading, not compiled, not run on hardware
//
#include <hip/hip_runtime.h>
#include <cstdint>
#include <cstddef>

#define D_MODEL 1024
#define SEQ_T   4096
#define BATCH   8
#define M_ROWS  (BATCH * SEQ_T)   // 32768
#define NCH     (BATCH * D_MODEL) // 8192 channels
#define NCHUNK  64
#define CHUNK_T 64                // NCHUNK * CHUNK_T == SEQ_T

typedef __attribute__((ext_vector_type(16))) __bf16 bf16x16;
typedef __attribute__((ext_vector_type(8)))  float  f32x8;

union FragB16 { unsigned int u[8]; bf16x16 v; };

__device__ __forceinline__ unsigned short f32_to_bf16_rne(float f) {
  union { float f; unsigned int u; } w; w.f = f;
  unsigned int r = w.u + 0x7FFFu + ((w.u >> 16) & 1u);
  return (unsigned short)(r >> 16);
}

// CDNA5 async global->LDS copy (ASYNCcnt-tracked). vdst = wave-relative LDS
// byte offset, vaddr = signed 32-bit byte offset, saddr = 64-bit base (GVS).
__device__ __forceinline__ void async_load_b128(unsigned lds_off, unsigned goff,
                                                unsigned long long base) {
  asm volatile("global_load_async_to_lds_b128 %0, %1, %2"
               :: "v"(lds_off), "v"(goff), "s"(base) : "memory");
}

// ---------------------------------------------------------------------------
// f32 -> bf16 conversion (4 elements / thread, packed 32-bit stores)
// ---------------------------------------------------------------------------
__global__ void cvt_f32_bf16_kernel(const float* __restrict__ in,
                                    unsigned short* __restrict__ out, int n) {
  int i = (blockIdx.x * blockDim.x + threadIdx.x) * 4;
  if (i >= n) return;
  float4 f = *(const float4*)&in[i];
  unsigned int p0 = ((unsigned int)f32_to_bf16_rne(f.y) << 16) | f32_to_bf16_rne(f.x);
  unsigned int p1 = ((unsigned int)f32_to_bf16_rne(f.w) << 16) | f32_to_bf16_rne(f.z);
  *(uint2*)&out[i] = make_uint2(p0, p1);
}

// ---------------------------------------------------------------------------
// lamb = exp(-(softplus(log_decay)+1e-4) * max(delta_t,1e-4))
// ---------------------------------------------------------------------------
__global__ void lamb_kernel(const float* __restrict__ delta_t,
                            const float* __restrict__ log_decay,
                            float* __restrict__ lamb) {
  size_t i = ((size_t)blockIdx.x * blockDim.x + threadIdx.x) * 4;
  int dbase = (int)(i & (D_MODEL - 1));
  float4 dt = *(const float4*)&delta_t[i];
  float4 ld = *(const float4*)&log_decay[dbase];
  float4 r;
  {
    float dec, t;
    dec = (ld.x > 20.f ? ld.x : log1pf(__expf(ld.x))) + 1e-4f;
    t = fmaxf(dt.x, 1e-4f); r.x = __expf(-dec * t);
    dec = (ld.y > 20.f ? ld.y : log1pf(__expf(ld.y))) + 1e-4f;
    t = fmaxf(dt.y, 1e-4f); r.y = __expf(-dec * t);
    dec = (ld.z > 20.f ? ld.z : log1pf(__expf(ld.z))) + 1e-4f;
    t = fmaxf(dt.z, 1e-4f); r.z = __expf(-dec * t);
    dec = (ld.w > 20.f ? ld.w : log1pf(__expf(ld.w))) + 1e-4f;
    t = fmaxf(dt.w, 1e-4f); r.w = __expf(-dec * t);
  }
  *(float4*)&lamb[i] = r;
}

// ---------------------------------------------------------------------------
// WMMA GEMM with async-to-LDS double buffering.
// out[m,n] = sum_k A[m,k]*W[n,k] (+ fused epilogue)
// Block tile 128x64, K-stage 64 (2 WMMA k-steps), 8 wave32s (4x2),
// 32x32 per wave (2x2 WMMA tiles). Stages prefetched with
// global_load_async_to_lds_b128, synchronized via s_wait_asynccnt.
// ---------------------------------------------------------------------------
template <bool DUAL>
__global__ __launch_bounds__(256)
void gemm_wmma_kernel(const unsigned short* __restrict__ A,    // [M, D] bf16
                      const unsigned short* __restrict__ B0,   // [D, D] bf16
                      const unsigned short* __restrict__ B1,   // [D, D] bf16 (DUAL)
                      const float* __restrict__ bias0,
                      const float* __restrict__ bias1,
                      const float* __restrict__ aux,           // DUAL: lamb; else residual
                      float* __restrict__ out)                 // [M, D]
{
  __shared__ __align__(16) unsigned short sA [2][128 * 64];            // 2 x 16 KB
  __shared__ __align__(16) unsigned short sB0[2][64 * 64];             // 2 x  8 KB
  __shared__ __align__(16) unsigned short sB1[DUAL ? 2 : 1][DUAL ? 64 * 64 : 16];

  const int tid  = threadIdx.x;
  const int lane = tid & 31;
  const int wave = tid >> 5;
  const int wm   = wave >> 1;     // 0..3  (M)
  const int wn   = wave & 1;      // 0..1  (N)
  const int lh   = lane >> 4;
  const int l16  = lane & 15;

  const int blockM = blockIdx.x * 128;
  const int blockN = blockIdx.y * 64;

  const unsigned long long baseA  = (unsigned long long)A;
  const unsigned long long baseB0 = (unsigned long long)B0;
  const unsigned long long baseB1 = (unsigned long long)B1;

  f32x8 acc0[2][2] = {};
  f32x8 acc1[2][2] = {};

  // Issue one K-stage (64 wide) of async copies for this wave's share.
  // A tile 128x64 bf16 = 1024 x b128 chunks -> 4 per lane.
  // B tiles 64x64 bf16 = 512 x b128 chunks  -> 2 per lane each.
  auto issue_stage = [&](int kk, int buf) {
#pragma unroll
    for (int i = 0; i < 4; ++i) {
      int c   = wave * 128 + i * 32 + lane;
      int row = c >> 3;
      int kp  = (c & 7) * 8;                       // bf16 units
      async_load_b128((unsigned)(uintptr_t)&sA[buf][row * 64 + kp],
                      (unsigned)(((size_t)(blockM + row) * D_MODEL + kk + kp) * 2),
                      baseA);
    }
#pragma unroll
    for (int i = 0; i < 2; ++i) {
      int c  = wave * 64 + i * 32 + lane;
      int n  = c >> 3;
      int kp = (c & 7) * 8;
      unsigned goff = (unsigned)(((size_t)(blockN + n) * D_MODEL + kk + kp) * 2);
      async_load_b128((unsigned)(uintptr_t)&sB0[buf][n * 64 + kp], goff, baseB0);
      if (DUAL)
        async_load_b128((unsigned)(uintptr_t)&sB1[buf][n * 64 + kp], goff, baseB1);
    }
  };

  const int NT = D_MODEL / 64;     // 16 stages
  issue_stage(0, 0);

#pragma unroll 1
  for (int it = 0; it < NT; ++it) {
    const int cur = it & 1;
    if (it + 1 < NT) {
      issue_stage((it + 1) * 64, cur ^ 1);
      // previous stage fully retired when only the just-issued batch remains
      if (DUAL) asm volatile("s_wait_asynccnt 8" ::: "memory");
      else      asm volatile("s_wait_asynccnt 6" ::: "memory");
    } else {
      asm volatile("s_wait_asynccnt 0" ::: "memory");
    }
    __syncthreads();

#pragma unroll
    for (int s = 0; s < 2; ++s) {               // two 16x16x32 k-steps
      FragB16 aF[2], b0F[2], b1F[2];
#pragma unroll
      for (int tm = 0; tm < 2; ++tm) {
        const unsigned short* pr = &sA[cur][(wm * 32 + tm * 16 + l16) * 64 + s * 32];
#pragma unroll
        for (int v = 0; v < 8; ++v) {
          int k = (v < 4 ? 0 : 16) + lh * 8 + (v & 3) * 2;
          aF[tm].u[v] = *(const unsigned int*)&pr[k];
        }
      }
#pragma unroll
      for (int tn = 0; tn < 2; ++tn) {
        const int colb = (wn * 32 + tn * 16 + l16) * 64 + s * 32;
#pragma unroll
        for (int v = 0; v < 8; ++v) {
          int k = (v < 4 ? 0 : 16) + lh * 8 + (v & 3) * 2;
          b0F[tn].u[v] = *(const unsigned int*)&sB0[cur][colb + k];
          if (DUAL) b1F[tn].u[v] = *(const unsigned int*)&sB1[cur][colb + k];
        }
      }
#pragma unroll
      for (int tm = 0; tm < 2; ++tm)
#pragma unroll
        for (int tn = 0; tn < 2; ++tn) {
          acc0[tm][tn] = __builtin_amdgcn_wmma_f32_16x16x32_bf16(
              false, aF[tm].v, false, b0F[tn].v, (short)0, acc0[tm][tn], false, false);
          if (DUAL)
            acc1[tm][tn] = __builtin_amdgcn_wmma_f32_16x16x32_bf16(
                false, aF[tm].v, false, b1F[tn].v, (short)0, acc1[tm][tn], false, false);
        }
    }
    __syncthreads();   // protect buffer before it is refilled next iteration
  }

  // ---- fused epilogue (C layout: lane holds N=l16, rows r+8*lh) ----
#pragma unroll
  for (int tm = 0; tm < 2; ++tm) {
#pragma unroll
    for (int tn = 0; tn < 2; ++tn) {
      int n = blockN + wn * 32 + tn * 16 + l16;
      float bv0 = bias0[n];
      float bv1 = DUAL ? bias1[n] : 0.0f;
#pragma unroll
      for (int r = 0; r < 8; ++r) {
        int m = blockM + wm * 32 + tm * 16 + lh * 8 + r;
        size_t idx = (size_t)m * D_MODEL + n;
        if (DUAL) {
          float u = acc0[tm][tn][r] + bv0;
          float g = acc1[tm][tn][r] + bv1;
          g = 1.0f / (1.0f + __expf(-g));
          float l = aux[idx];
          out[idx] = (1.0f - l) * g * u;
        } else {
          out[idx] = acc0[tm][tn][r] + bv0 + aux[idx];
        }
      }
    }
  }
}

// ---------------------------------------------------------------------------
// Chunked affine scan.  Step s -> lamb*s + drive composes associatively:
//   (a2,b2) o (a1,b1) = (a2*a1, a2*b1 + b2)
// Phase A: per (chunk, channel) fold 64 steps -> (A_c, B_c).
// Phase B: per channel scan 64 summaries -> chunk-start states + final_state.
// Phase C: per (chunk, channel) replay chunk from start state, write states.
// ---------------------------------------------------------------------------
__global__ void scan_phaseA_kernel(const float* __restrict__ lamb,
                                   const float* __restrict__ drive,
                                   float2* __restrict__ cs) {
  int gid = blockIdx.x * blockDim.x + threadIdx.x;   // 0 .. NCHUNK*NCH-1
  int c   = gid >> 13;                               // chunk
  int ch  = gid & (NCH - 1);
  int b   = ch >> 10;
  size_t base = (size_t)b * SEQ_T * D_MODEL + (ch & (D_MODEL - 1))
              + (size_t)(c * CHUNK_T) * D_MODEL;
  float a = 1.0f, bb = 0.0f;
#pragma unroll 4
  for (int t = 0; t < CHUNK_T; ++t) {
    size_t i = base + (size_t)t * D_MODEL;
    float l = lamb[i], dr = drive[i];
    a *= l;
    bb = fmaf(l, bb, dr);
  }
  cs[gid] = make_float2(a, bb);
}

__global__ void scan_phaseB_kernel(const float2* __restrict__ cs,
                                   float* __restrict__ start,
                                   float* __restrict__ final_state) {
  int ch = blockIdx.x * blockDim.x + threadIdx.x;    // 0..NCH-1
  float s = 0.0f;
#pragma unroll 1
  for (int c = 0; c < NCHUNK; ++c) {
    start[c * NCH + ch] = s;
    float2 ab = cs[c * NCH + ch];
    s = fmaf(ab.x, s, ab.y);
  }
  final_state[ch] = s;
}

__global__ void scan_phaseC_kernel(float* __restrict__ lamb_states,   // in-place
                                   const float* __restrict__ drive,
                                   const float* __restrict__ start,
                                   unsigned short* __restrict__ states_bf) {
  int gid = blockIdx.x * blockDim.x + threadIdx.x;
  int c   = gid >> 13;
  int ch  = gid & (NCH - 1);
  int b   = ch >> 10;
  size_t base = (size_t)b * SEQ_T * D_MODEL + (ch & (D_MODEL - 1))
              + (size_t)(c * CHUNK_T) * D_MODEL;
  float s = start[c * NCH + ch];
#pragma unroll 4
  for (int t = 0; t < CHUNK_T; ++t) {
    size_t i = base + (size_t)t * D_MODEL;
    float l = lamb_states[i], dr = drive[i];
    s = fmaf(l, s, dr);
    lamb_states[i] = s;
    states_bf[i]   = f32_to_bf16_rne(s);
  }
}

// ---------------------------------------------------------------------------
// In-place LayerNorm over D=1024 per row; 256 threads, 4 elems/thread.
// ---------------------------------------------------------------------------
__global__ __launch_bounds__(256)
void layernorm_kernel(float* __restrict__ y,
                      const float* __restrict__ gamma,
                      const float* __restrict__ beta) {
  __shared__ float ss[8], ssq[8];
  int row = blockIdx.x;
  int tid = threadIdx.x;
  float* py = y + (size_t)row * D_MODEL;

  float v[4];
  float s = 0.f, sq = 0.f;
#pragma unroll
  for (int i = 0; i < 4; ++i) {
    v[i] = py[tid + i * 256];
    s += v[i];
    sq += v[i] * v[i];
  }
#pragma unroll
  for (int off = 16; off > 0; off >>= 1) {
    s  += __shfl_xor(s,  off, 32);
    sq += __shfl_xor(sq, off, 32);
  }
  int wv = tid >> 5, lane = tid & 31;
  if (lane == 0) { ss[wv] = s; ssq[wv] = sq; }
  __syncthreads();
  float ts = 0.f, tsq = 0.f;
#pragma unroll
  for (int w = 0; w < 8; ++w) { ts += ss[w]; tsq += ssq[w]; }
  float mu   = ts * (1.0f / D_MODEL);
  float var  = tsq * (1.0f / D_MODEL) - mu * mu;
  float rstd = rsqrtf(var + 1e-5f);
#pragma unroll
  for (int i = 0; i < 4; ++i) {
    int d = tid + i * 256;
    py[d] = (v[i] - mu) * rstd * gamma[d] + beta[d];
  }
}

// ---------------------------------------------------------------------------
extern "C" void kernel_launch(void* const* d_in, const int* in_sizes, int n_in,
                              void* d_out, int out_size, void* d_ws, size_t ws_size,
                              hipStream_t stream) {
  const float* x         = (const float*)d_in[0];
  const float* delta_t   = (const float*)d_in[1];
  const float* log_decay = (const float*)d_in[2];
  const float* W_in      = (const float*)d_in[3];
  const float* b_in      = (const float*)d_in[4];
  const float* W_gate    = (const float*)d_in[5];
  const float* b_gate    = (const float*)d_in[6];
  const float* W_out     = (const float*)d_in[7];
  const float* b_out     = (const float*)d_in[8];
  const float* gamma     = (const float*)d_in[9];
  const float* beta      = (const float*)d_in[10];

  const size_t MD = (size_t)M_ROWS * D_MODEL;
  const size_t DD = (size_t)D_MODEL * D_MODEL;

  char* ws = (char*)d_ws;
  size_t off = 0;
  auto alloc = [&](size_t bytes) {
    char* p = ws + off;
    off += (bytes + 255) & ~(size_t)255;
    return p;
  };
  unsigned short* xb        = (unsigned short*)alloc(MD * 2);           //  64 MB
  unsigned short* wb_in     = (unsigned short*)alloc(DD * 2);           //   2 MB
  unsigned short* wb_gate   = (unsigned short*)alloc(DD * 2);           //   2 MB
  unsigned short* wb_out    = (unsigned short*)alloc(DD * 2);           //   2 MB
  float*          lamb_st   = (float*)alloc(MD * 4);                    // 128 MB
  float*          drive     = (float*)alloc(MD * 4);                    // 128 MB
  unsigned short* states_bf = (unsigned short*)alloc(MD * 2);           //  64 MB
  float2*         csum      = (float2*)alloc((size_t)NCHUNK * NCH * 8); //   4 MB
  float*          cstart    = (float*)alloc((size_t)NCHUNK * NCH * 4);  //   2 MB

  float* out_y       = (float*)d_out;   // [M, D], normalized in place
  float* final_state = out_y + MD;      // [B, D]

  // 1) conversions to bf16
  cvt_f32_bf16_kernel<<<(int)(MD / 4 / 256), 256, 0, stream>>>(x, xb, (int)MD);
  cvt_f32_bf16_kernel<<<(int)(DD / 4 / 256), 256, 0, stream>>>(W_in, wb_in, (int)DD);
  cvt_f32_bf16_kernel<<<(int)(DD / 4 / 256), 256, 0, stream>>>(W_gate, wb_gate, (int)DD);
  cvt_f32_bf16_kernel<<<(int)(DD / 4 / 256), 256, 0, stream>>>(W_out, wb_out, (int)DD);

  // 2) lamb
  lamb_kernel<<<(int)(MD / 4 / 256), 256, 0, stream>>>(delta_t, log_decay, lamb_st);

  // 3) fused dual GEMM (update & gate) + drive epilogue
  dim3 gemm_grid(M_ROWS / 128, D_MODEL / 64);
  gemm_wmma_kernel<true><<<gemm_grid, 256, 0, stream>>>(
      xb, wb_in, wb_gate, b_in, b_gate, lamb_st, drive);

  // 4) chunk-parallel scan
  scan_phaseA_kernel<<<(NCHUNK * NCH) / 256, 256, 0, stream>>>(lamb_st, drive, csum);
  scan_phaseB_kernel<<<NCH / 256, 256, 0, stream>>>(csum, cstart, final_state);
  scan_phaseC_kernel<<<(NCHUNK * NCH) / 256, 256, 0, stream>>>(
      lamb_st, drive, cstart, states_bf);

  // 5) output GEMM: y = states @ W_out^T + b_out + states -> d_out
  gemm_wmma_kernel<false><<<gemm_grid, 256, 0, stream>>>(
      states_bf, wb_out, wb_out, b_out, b_out, lamb_st, out_y);

  // 6) in-place LayerNorm
  layernorm_kernel<<<M_ROWS, 256, 0, stream>>>(out_y, gamma, beta);

  (void)in_sizes; (void)n_in; (void)out_size; (void)ws_size;
}
